// ContrastiveLoss_87050397155711
// MI455X (gfx1250) — matrix operand order repaired and checked by reading, not compiled, not run on hardware
//
#include <hip/hip_runtime.h>

typedef __attribute__((ext_vector_type(16))) _Float16 v16h;
typedef __attribute__((ext_vector_type(8)))  _Float16 v8h;
typedef __attribute__((ext_vector_type(8)))  float    v8f;

#define B_ROWS  4096
#define DIM     256
#define NROWS   8192      // 2*B
#define NTILES  512       // NROWS / 16
#define INV_T   5.0f      // 1 / 0.2
#define T_LOG2E 7.2134752044448170368f   // 5 * log2(e):  exp(5x) == exp2(x * this)

// Raw hardware exp2 (v_exp_f32). Our arguments are in [-7.3, 7.3], far from
// the -126 denorm threshold, so the OCML denorm-safe expansion is not needed.
#if __has_builtin(__builtin_amdgcn_exp2f)
#define FAST_EXP2(x) __builtin_amdgcn_exp2f(x)
#else
#define FAST_EXP2(x) exp2f(x)
#endif

// ---------------------------------------------------------------------------
// Phase 1: L2-normalize each row of [emb_i; emb_j] (fp32) -> z (f16, row-major
// [8192][256]) in workspace. One wave32 per row: 8 floats per lane.
// ---------------------------------------------------------------------------
__global__ void ntxent_normalize_kernel(const float* __restrict__ emb_i,
                                        const float* __restrict__ emb_j,
                                        _Float16* __restrict__ z) {
  const int wave = threadIdx.x >> 5;
  const int lane = threadIdx.x & 31;
  const int row  = blockIdx.x * 8 + wave;          // 1024 blocks * 8 waves = 8192

  const float* src = (row < B_ROWS) ? (emb_i + (size_t)row * DIM)
                                    : (emb_j + (size_t)(row - B_ROWS) * DIM);
  float4 p0 = *(const float4*)(src + lane * 8);
  float4 p1 = *(const float4*)(src + lane * 8 + 4);

  float ss = p0.x*p0.x + p0.y*p0.y + p0.z*p0.z + p0.w*p0.w
           + p1.x*p1.x + p1.y*p1.y + p1.z*p1.z + p1.w*p1.w;
  ss += __shfl_xor(ss, 16, 32);
  ss += __shfl_xor(ss,  8, 32);
  ss += __shfl_xor(ss,  4, 32);
  ss += __shfl_xor(ss,  2, 32);
  ss += __shfl_xor(ss,  1, 32);
  const float rn = rsqrtf(ss);

  v8h o;
  o[0] = (_Float16)(p0.x * rn); o[1] = (_Float16)(p0.y * rn);
  o[2] = (_Float16)(p0.z * rn); o[3] = (_Float16)(p0.w * rn);
  o[4] = (_Float16)(p1.x * rn); o[5] = (_Float16)(p1.y * rn);
  o[6] = (_Float16)(p1.z * rn); o[7] = (_Float16)(p1.w * rn);
  *(v8h*)(z + (size_t)row * DIM + lane * 8) = o;
}

// ---------------------------------------------------------------------------
// Phase 2: fused sim = z@z^T sweep. 8 waves/block, one 16-row strip per wave.
// All 8 waves consume the SAME column tile each iteration, so the 8 KB B tile
// (16 rows x 256 f16, contiguous in memory) is staged once per block into a
// double-buffered LDS stage with CDNA5 async-to-LDS loads (ASYNCcnt), then
// each wave feeds its 8 chained v_wmma_f32_16x16x32_f16 from LDS. This cuts
// L2-side B traffic 8x (2 GB -> 256 MB) and overlaps the next tile's DMA with
// the current tile's WMMAs.
// C/D layout: VGPR i <-> (M=i, N=lane) lanes 0-15, (M=8+i, N=lane-16) lanes
// 16-31; diagonal / positive element sits at lane==i or lane==24+i.
// Epilogue is kept strictly select-form (no EXEC branches) so the per-tile
// cost is 8x {v_mul, v_exp_f32, v_cndmask, v_add}.
// ---------------------------------------------------------------------------
__global__ void __launch_bounds__(256)
ntxent_sim_kernel(const _Float16* __restrict__ z,
                  float* __restrict__ tile_partials) {
  __shared__ __align__(32) _Float16 bstage[2 * 16 * DIM];   // 16 KB double buffer

  const int tid  = threadIdx.x;
  const int wave = tid >> 5;
  const int lane = tid & 31;
  const int rt   = blockIdx.x * 8 + wave;          // 64 blocks * 8 waves = 512
  const int r0   = rt * 16;
  const int m    = lane & 15;                      // A row / B col within tile
  const int kh   = lane >> 4;                      // K half-select

  // A fragments (rows r0..r0+15, K=0..255), resident for the whole sweep.
  v16h a[8];
  {
    const _Float16* za = z + (size_t)(r0 + m) * DIM;
#pragma unroll
    for (int kc = 0; kc < 8; ++kc) {
      v8h lo = *(const v8h*)(za + kc * 32 + kh * 16);
      v8h hi = *(const v8h*)(za + kc * 32 + kh * 16 + 8);
#pragma unroll
      for (int j = 0; j < 8; ++j) { a[kc][j] = lo[j]; a[kc][8 + j] = hi[j]; }
    }
  }

  const unsigned long long zaddr = (unsigned long long)(uintptr_t)z;
  const unsigned lbase = (unsigned)(uintptr_t)(&bstage[0]);  // LDS byte address
  const unsigned stoff = (unsigned)tid * 16u;   // 256 thr * 2 * 16B = 8 KB/tile

  // Prologue: stage column tile 0 into buffer 0.
  {
    unsigned l = lbase + stoff;
    unsigned g = stoff;
    asm volatile("global_load_async_to_lds_b128 %0, %1, %2 offset:0\n\t"
                 "global_load_async_to_lds_b128 %0, %1, %2 offset:4096"
                 :: "v"(l), "v"(g), "s"(zaddr) : "memory");
  }
  asm volatile("s_wait_asynccnt 0x0" ::: "memory");
  __syncthreads();

  float sumExp[8];
  float posv[8];
#pragma unroll
  for (int i = 0; i < 8; ++i) { sumExp[i] = 0.0f; posv[i] = 0.0f; }

  const int posct = rt ^ 256;                      // col tile holding sim[r, r +/- B]

  for (int ct = 0; ct < NTILES; ++ct) {
    const int buf = ct & 1;

    // Kick off async DMA of the next tile into the other buffer (overlaps WMMA).
    if (ct + 1 < NTILES) {
      unsigned l = lbase + (unsigned)(buf ^ 1) * 8192u + stoff;
      unsigned g = (unsigned)(ct + 1) * 8192u + stoff;
      asm volatile("global_load_async_to_lds_b128 %0, %1, %2 offset:0\n\t"
                   "global_load_async_to_lds_b128 %0, %1, %2 offset:4096"
                   :: "v"(l), "v"(g), "s"(zaddr) : "memory");
    }

    // B fragments from LDS: halves 0..15 <-> K = 32*kc + 16*kh + (0..15).
    const _Float16* bb = bstage + buf * (16 * DIM) + m * DIM;
    v8f c = {};
#pragma unroll
    for (int kc = 0; kc < 8; ++kc) {
      v16h b = *(const v16h*)(bb + kc * 32 + kh * 16);
      c = __builtin_amdgcn_wmma_f32_16x16x32_f16(
              /*neg_a=*/false, a[kc], /*neg_b=*/false, b,
              /*c_mod=*/(short)0, c, /*reuse_a=*/false, /*reuse_b=*/false);
    }

    const bool isDiagTile = (ct == rt);
    const bool isPosTile  = (ct == posct);
#pragma unroll
    for (int i = 0; i < 8; ++i) {
      const float v = c[i];
      const bool onDiagLane = (lane == i) || (lane == 24 + i);
      const float e = FAST_EXP2(v * T_LOG2E);
      // denom excludes the self-similarity (main diagonal of sim)
      sumExp[i] += (isDiagTile && onDiagLane) ? 0.0f : e;
      posv[i]    = (isPosTile  && onDiagLane) ? v : posv[i];
    }

    // My async ops for the next tile are done; barrier makes them visible to
    // all waves and frees the buffer everyone just finished reading.
    asm volatile("s_wait_asynccnt 0x0" ::: "memory");
    __syncthreads();
  }

  // Row sums: reduce each accumulator element across its 16-lane group, then
  // the lane that owns the row's positive computes log(denom) - pos/T.
  float partial = 0.0f;
#pragma unroll
  for (int i = 0; i < 8; ++i) {
    float s = sumExp[i];
    s += __shfl_xor(s, 8, 32);
    s += __shfl_xor(s, 4, 32);
    s += __shfl_xor(s, 2, 32);
    s += __shfl_xor(s, 1, 32);
    if ((lane == i) || (lane == 24 + i)) partial += logf(s) - posv[i] * INV_T;
  }
  partial += __shfl_xor(partial, 16, 32);
  partial += __shfl_xor(partial,  8, 32);
  partial += __shfl_xor(partial,  4, 32);
  partial += __shfl_xor(partial,  2, 32);
  partial += __shfl_xor(partial,  1, 32);
  if (lane == 0) tile_partials[rt] = partial;
}

// ---------------------------------------------------------------------------
// Phase 3: deterministic fixed-order reduction of 512 tile partials -> loss.
// ---------------------------------------------------------------------------
__global__ void ntxent_reduce_kernel(const float* __restrict__ tile_partials,
                                     float* __restrict__ out) {
  const int lane = threadIdx.x;                    // 32 threads
  float s = 0.0f;
  for (int i = lane; i < NTILES; i += 32) s += tile_partials[i];
  s += __shfl_xor(s, 16, 32);
  s += __shfl_xor(s,  8, 32);
  s += __shfl_xor(s,  4, 32);
  s += __shfl_xor(s,  2, 32);
  s += __shfl_xor(s,  1, 32);
  if (lane == 0) out[0] = s * (1.0f / (float)NROWS);
}

extern "C" void kernel_launch(void* const* d_in, const int* in_sizes, int n_in,
                              void* d_out, int out_size, void* d_ws, size_t ws_size,
                              hipStream_t stream) {
  const float* emb_i = (const float*)d_in[0];
  const float* emb_j = (const float*)d_in[1];
  float* out = (float*)d_out;

  _Float16* z = (_Float16*)d_ws;                                        // 4 MB
  float* tile_partials = (float*)((char*)d_ws + (size_t)NROWS * DIM * 2); // 2 KB

  ntxent_normalize_kernel<<<NROWS / 8, 256, 0, stream>>>(emb_i, emb_j, z);
  ntxent_sim_kernel<<<NTILES / 8, 256, 0, stream>>>(z, tile_partials);
  ntxent_reduce_kernel<<<1, 32, 0, stream>>>(tile_partials, out);
}
